// Chambers_16578573762686
// MI455X (gfx1250) — matrix operand order repaired and checked by reading, not compiled, not run on hardware
//
#include <hip/hip_runtime.h>
#include <hip/hip_bf16.h>

typedef __attribute__((ext_vector_type(16))) __bf16 v16bf;
typedef __attribute__((ext_vector_type(8)))  __bf16 v8bf;
typedef __attribute__((ext_vector_type(8)))  float  v8f;

#define B_TOTAL 131072
#define RES_DIM 100
#define NCH     6
#define ROWS    128
#define NTHR    256
#define NWAVES  8
#define CF_K    0.02f

// Row pitches for activations (padded +8 elements to break LDS bank conflicts:
// 136*2B = 68 DWORDs, so 16 consecutive rows land on distinct 4-bank groups).
#define LDX  136
#define LDH1 136
#define LDH2 72
#define LDH3 40

// LDS element counts
#define SW1_N (128*128)     // W1 frag-major, K padded 100->128
#define SW2_N (128*64)      // W2 frag-major
#define SW3_N (64*32)       // W3 frag-major
#define SX_N  (128*LDX)
#define SH1_N (128*LDH1)
#define SH2_N (128*LDH2)
#define SH3_N (128*LDH3)
#define BF16_TOTAL (SW1_N+SW2_N+SW3_N+SX_N+SH1_N+SH2_N+SH3_N)
#define SMEM_BYTES (BF16_TOTAL*2 + (128+64+32+32+1)*4)

// Fast activation math: v_exp_f32 + v_rcp_f32 (~1 ulp) instead of the IEEE
// divide expansion (div_scale/rcp/2xfma/div_fmas/div_fixup). Far more accurate
// than the bf16 storage that follows.
__device__ __forceinline__ float silu_f(float x)
{
    return x * __builtin_amdgcn_rcpf(1.f + __expf(-x));
}
__device__ __forceinline__ float sigm_f(float x)
{
    return __builtin_amdgcn_rcpf(1.f + __expf(-x));
}

// A fragment for one 16x32 k-step of a row-major [M][K] LDS matrix.
// Per CDNA5 ISA 7.12.2 (16-bit A 16x32): lane l<16 holds row M=l, K {0..7,16..23};
// lane l>=16 holds row M=l-16, K {8..15,24..31}. Two aligned 16B LDS loads.
__device__ __forceinline__ v16bf load_a_frag(const __bf16* __restrict__ A, int lda,
                                             int mt, int ks, int l15, int hi)
{
    const __bf16* Ap = A + (mt * 16 + l15) * lda + ks * 32 + hi * 8;
    v8bf lo  = *(const v8bf*)(Ap);
    v8bf hi8 = *(const v8bf*)(Ap + 16);
    return __builtin_shufflevector(lo, hi8, 0,1,2,3,4,5,6,7,8,9,10,11,12,13,14,15);
}

// B fragments live in LDS pre-swizzled to fragment-major layout:
//   offset = (((nt*KSTEPS)+ks)*32 + lane)*16 + j
// so each lane reads its 16 bf16 contiguously (32B aligned -> 2x ds_load_b128).
__device__ __forceinline__ v16bf load_b_frag(const __bf16* __restrict__ Wf,
                                             int ksteps, int nt, int ks, int lane)
{
    return *(const v16bf*)(Wf + (((nt * ksteps) + ks) * 32 + lane) * 16);
}

__global__ void __launch_bounds__(NTHR)
chambers_mlp_kernel(const float* __restrict__ res,
                    const float* __restrict__ W1, const float* __restrict__ b1,
                    const float* __restrict__ W2, const float* __restrict__ b2,
                    const float* __restrict__ W3, const float* __restrict__ b3,
                    const float* __restrict__ W4, const float* __restrict__ b4,
                    float* __restrict__ raw_out)
{
    extern __shared__ char smem[];
    __bf16* sW1f = (__bf16*)smem;          // frag-major, 32B-aligned regions
    __bf16* sW2f = sW1f + SW1_N;
    __bf16* sW3f = sW2f + SW2_N;
    __bf16* sX   = sW3f + SW3_N;           // row-major [128][LDX]
    __bf16* sH1  = sX   + SX_N;            // [128][LDH1]
    __bf16* sH2  = sH1  + SH1_N;           // [128][LDH2]
    __bf16* sH3  = sH2  + SH2_N;           // [128][LDH3]
    float*  sB1f = (float*)(sH3 + SH3_N);
    float*  sB2f = sB1f + 128;
    float*  sB3f = sB2f + 64;
    float*  sW4f = sB3f + 32;
    float*  sB4f = sW4f + 32;

    const int c    = blockIdx.y;
    const int row0 = blockIdx.x * ROWS;
    const int tid  = threadIdx.x;

    // ---- Stage W1 into fragment-major layout (K padded 100->128, KSTEPS=4, NT=8)
    for (int i = tid; i < SW1_N; i += NTHR) {
        int j  = i & 15;
        int ln = (i >> 4) & 31;
        int ks = (i >> 9) & 3;
        int nt = i >> 11;
        int k  = ks * 32 + ((ln >> 4) << 3) + (j & 7) + ((j >> 3) << 4);
        int n  = (nt << 4) + (ln & 15);
        float v = (k < RES_DIM) ? W1[((size_t)c * RES_DIM + k) * 128 + n] : 0.f;
        sW1f[i] = (__bf16)v;
    }
    // ---- Stage W2 (K=128, KSTEPS=4, NT=4)
    for (int i = tid; i < SW2_N; i += NTHR) {
        int j  = i & 15;
        int ln = (i >> 4) & 31;
        int ks = (i >> 9) & 3;
        int nt = i >> 11;
        int k  = ks * 32 + ((ln >> 4) << 3) + (j & 7) + ((j >> 3) << 4);
        int n  = (nt << 4) + (ln & 15);
        sW2f[i] = (__bf16)W2[((size_t)c * 128 + k) * 64 + n];
    }
    // ---- Stage W3 (K=64, KSTEPS=2, NT=2)
    for (int i = tid; i < SW3_N; i += NTHR) {
        int j  = i & 15;
        int ln = (i >> 4) & 31;
        int ks = (i >> 9) & 1;
        int nt = i >> 10;
        int k  = ks * 32 + ((ln >> 4) << 3) + (j & 7) + ((j >> 3) << 4);
        int n  = (nt << 4) + (ln & 15);
        sW3f[i] = (__bf16)W3[((size_t)c * 64 + k) * 32 + n];
    }
    if (tid < 128) sB1f[tid] = b1[c * 128 + tid];
    if (tid < 64)  sB2f[tid] = b2[c * 64 + tid];
    if (tid < 32)  sB3f[tid] = b3[c * 32 + tid];
    if (tid >= 32 && tid < 64) sW4f[tid - 32] = W4[c * 32 + (tid - 32)];
    if (tid == NTHR - 1) sB4f[0] = b4[c];

    // ---- Stage res tile row-major with padded pitch (zero-pad K 100->128)
    for (int i = tid; i < 128 * 128; i += NTHR) {
        int r = i >> 7, k = i & 127;
        float v = (k < RES_DIM) ? res[(size_t)(row0 + r) * RES_DIM + k] : 0.f;
        sX[r * LDX + k] = (__bf16)v;
    }
    __syncthreads();

    const int wave = tid >> 5;
    const int lane = tid & 31;
    const int l15  = lane & 15;
    const int hi   = lane >> 4;

    // ======== Layer 1: h1 = silu(X[128x128] @ W1[128x128] + b1) ========
    // nt fixed per wave (8 ntiles / 8 waves); B frags loaded once, held in regs.
    {
        const int nt = wave;
        v16bf bf0 = load_b_frag(sW1f, 4, nt, 0, lane);
        v16bf bf1 = load_b_frag(sW1f, 4, nt, 1, lane);
        v16bf bf2 = load_b_frag(sW1f, 4, nt, 2, lane);
        v16bf bf3 = load_b_frag(sW1f, 4, nt, 3, lane);
        const int n = nt * 16 + l15;
        const float bias = sB1f[n];
        for (int mt = 0; mt < 8; ++mt) {
            // Issue all A loads up front so DS traffic overlaps the WMMA chain.
            v16bf a0 = load_a_frag(sX, LDX, mt, 0, l15, hi);
            v16bf a1 = load_a_frag(sX, LDX, mt, 1, l15, hi);
            v16bf a2 = load_a_frag(sX, LDX, mt, 2, l15, hi);
            v16bf a3 = load_a_frag(sX, LDX, mt, 3, l15, hi);
            v8f acc = {};
            acc = __builtin_amdgcn_wmma_f32_16x16x32_bf16(false, a0, false, bf0, (short)0, acc, false, false);
            acc = __builtin_amdgcn_wmma_f32_16x16x32_bf16(false, a1, false, bf1, (short)0, acc, false, false);
            acc = __builtin_amdgcn_wmma_f32_16x16x32_bf16(false, a2, false, bf2, (short)0, acc, false, false);
            acc = __builtin_amdgcn_wmma_f32_16x16x32_bf16(false, a3, false, bf3, (short)0, acc, false, false);
#pragma unroll
            for (int r = 0; r < 8; ++r) {
                int m = mt * 16 + r + hi * 8;   // C layout: M = vgpr + 8*halfwave
                sH1[m * LDH1 + n] = (__bf16)silu_f(acc[r] + bias);
            }
        }
    }
    __syncthreads();

    // ======== Layer 2: h2 = silu(h1[128x128] @ W2[128x64] + b2) ========
    // 4 ntiles: nt = wave&3, two waves split the 8 mtiles.
    {
        const int nt = wave & 3;
        v16bf bf0 = load_b_frag(sW2f, 4, nt, 0, lane);
        v16bf bf1 = load_b_frag(sW2f, 4, nt, 1, lane);
        v16bf bf2 = load_b_frag(sW2f, 4, nt, 2, lane);
        v16bf bf3 = load_b_frag(sW2f, 4, nt, 3, lane);
        const int n = nt * 16 + l15;
        const float bias = sB2f[n];
        for (int mt = wave >> 2; mt < 8; mt += 2) {
            v16bf a0 = load_a_frag(sH1, LDH1, mt, 0, l15, hi);
            v16bf a1 = load_a_frag(sH1, LDH1, mt, 1, l15, hi);
            v16bf a2 = load_a_frag(sH1, LDH1, mt, 2, l15, hi);
            v16bf a3 = load_a_frag(sH1, LDH1, mt, 3, l15, hi);
            v8f acc = {};
            acc = __builtin_amdgcn_wmma_f32_16x16x32_bf16(false, a0, false, bf0, (short)0, acc, false, false);
            acc = __builtin_amdgcn_wmma_f32_16x16x32_bf16(false, a1, false, bf1, (short)0, acc, false, false);
            acc = __builtin_amdgcn_wmma_f32_16x16x32_bf16(false, a2, false, bf2, (short)0, acc, false, false);
            acc = __builtin_amdgcn_wmma_f32_16x16x32_bf16(false, a3, false, bf3, (short)0, acc, false, false);
#pragma unroll
            for (int r = 0; r < 8; ++r) {
                int m = mt * 16 + r + hi * 8;
                sH2[m * LDH2 + n] = (__bf16)silu_f(acc[r] + bias);
            }
        }
    }
    __syncthreads();

    // ======== Layer 3: h3 = silu(h2[128x64] @ W3[64x32] + b3) ========
    // 2 ntiles: nt = wave&1, four waves split the 8 mtiles.
    {
        const int nt = wave & 1;
        v16bf bf0 = load_b_frag(sW3f, 2, nt, 0, lane);
        v16bf bf1 = load_b_frag(sW3f, 2, nt, 1, lane);
        const int n = nt * 16 + l15;
        const float bias = sB3f[n];
        for (int mt = wave >> 1; mt < 8; mt += 4) {
            v16bf a0 = load_a_frag(sH2, LDH2, mt, 0, l15, hi);
            v16bf a1 = load_a_frag(sH2, LDH2, mt, 1, l15, hi);
            v8f acc = {};
            acc = __builtin_amdgcn_wmma_f32_16x16x32_bf16(false, a0, false, bf0, (short)0, acc, false, false);
            acc = __builtin_amdgcn_wmma_f32_16x16x32_bf16(false, a1, false, bf1, (short)0, acc, false, false);
#pragma unroll
            for (int r = 0; r < 8; ++r) {
                int m = mt * 16 + r + hi * 8;
                sH3[m * LDH3 + n] = (__bf16)silu_f(acc[r] + bias);
            }
        }
    }
    __syncthreads();

    // ======== Layer 4: raw[b, c] = h3 . W4 + b4 (one row per thread) ========
    if (tid < ROWS) {
        const v8bf* p = (const v8bf*)(sH3 + tid * LDH3);   // 16B-aligned chunks
        float acc = sB4f[0];
#pragma unroll
        for (int q = 0; q < 4; ++q) {
            v8bf v = p[q];
#pragma unroll
            for (int j = 0; j < 8; ++j)
                acc += (float)v[j] * sW4f[q * 8 + j];
        }
        raw_out[(size_t)(row0 + tid) * NCH + c] = acc;
    }
}

// Coupled sigmoid fixed-point: one thread per batch row; 6x6 coupling in
// registers with decay and CF_K pre-folded.
__global__ void __launch_bounds__(256)
chambers_couple_kernel(const float* __restrict__ raw,
                       const float* __restrict__ coupling,
                       const float* __restrict__ decay,
                       float* __restrict__ act)
{
    int b = blockIdx.x * 256 + threadIdx.x;
    if (b >= B_TOTAL) return;

    float Kd[NCH][NCH];
#pragma unroll
    for (int i = 0; i < NCH; ++i) {
        float di = decay[i] * CF_K;
#pragma unroll
        for (int j = 0; j < NCH; ++j)
            Kd[i][j] = di * coupling[i * NCH + j];
    }

    float r[NCH], a[NCH];
#pragma unroll
    for (int c0 = 0; c0 < NCH; ++c0) {
        r[c0] = raw[(size_t)b * NCH + c0];
        a[c0] = sigm_f(r[c0]);
    }
    for (int it = 0; it < 5; ++it) {
        float d[NCH];
#pragma unroll
        for (int j = 0; j < NCH; ++j) {
            float s = 0.f;
#pragma unroll
            for (int i = 0; i < NCH; ++i) s += a[i] * Kd[i][j];
            d[j] = s;
        }
#pragma unroll
        for (int j = 0; j < NCH; ++j) a[j] = sigm_f(r[j] + d[j]);
    }
#pragma unroll
    for (int c0 = 0; c0 < NCH; ++c0)
        act[(size_t)b * NCH + c0] = a[c0];
}

extern "C" void kernel_launch(void* const* d_in, const int* in_sizes, int n_in,
                              void* d_out, int out_size, void* d_ws, size_t ws_size,
                              hipStream_t stream)
{
    const float* res      = (const float*)d_in[0];
    const float* W1       = (const float*)d_in[1];
    const float* b1       = (const float*)d_in[2];
    const float* W2       = (const float*)d_in[3];
    const float* b2       = (const float*)d_in[4];
    const float* W3       = (const float*)d_in[5];
    const float* b3       = (const float*)d_in[6];
    const float* W4       = (const float*)d_in[7];
    const float* b4       = (const float*)d_in[8];
    const float* coupling = (const float*)d_in[9];
    const float* decay    = (const float*)d_in[10];

    float* act = (float*)d_out;                  // (B, 6)
    float* raw = act + (size_t)B_TOTAL * NCH;    // (B, 6), second tuple element

    static_assert(SMEM_BYTES < 320 * 1024, "exceeds WGP LDS");
    (void)hipFuncSetAttribute(reinterpret_cast<const void*>(chambers_mlp_kernel),
                              hipFuncAttributeMaxDynamicSharedMemorySize, SMEM_BYTES);

    dim3 grid(B_TOTAL / ROWS, NCH);
    chambers_mlp_kernel<<<grid, NTHR, SMEM_BYTES, stream>>>(
        res, W1, b1, W2, b2, W3, b3, W4, b4, raw);

    chambers_couple_kernel<<<(B_TOTAL + 255) / 256, 256, 0, stream>>>(
        raw, coupling, decay, act);
}
